// MultiResKAPatchEmbedding_65558380806622
// MI455X (gfx1250) — compile-verified, matrix operand order checked by reading, not run on hardware
//
#include <hip/hip_runtime.h>

// ---------------------------------------------------------------------------
// MI455X (gfx1250) implementation of MultiResKAPatchEmbedding.
// Compute-bound GEMM pipeline: v_wmma_f32_16x16x32_f16, async LDS staging,
// double-buffered tiles, wave tile 64x32 (8 WMMA / k-step / wave).
// ---------------------------------------------------------------------------

typedef _Float16 half_t;
typedef __attribute__((ext_vector_type(16))) _Float16 v16h;
typedef __attribute__((ext_vector_type(8)))  _Float16 v8h;
typedef __attribute__((ext_vector_type(8)))  float    v8f;
typedef __attribute__((ext_vector_type(4)))  int      v4i;

#define LDSS 40   // halves per LDS row: 32 data + 8 pad, keeps 16B alignment
#define BM   256
#define BN   64
#define BK   32

#if __has_builtin(__builtin_amdgcn_global_load_async_to_lds_b128) && \
    __has_builtin(__builtin_amdgcn_s_wait_asynccnt)
#define USE_ASYNC 1
#else
#define USE_ASYNC 0
#endif

#define AS1 __attribute__((address_space(1)))
#define AS3 __attribute__((address_space(3)))

#if USE_ASYNC
// Cast helpers via integer round-trip (always-legal conversions; LDS generic
// addresses carry the byte offset in the low 32 bits per ISA 10.2).
__device__ __forceinline__ AS1 v4i* as_global_v4i(const void* p) {
  return (AS1 v4i*)(unsigned long long)p;
}
__device__ __forceinline__ AS3 v4i* as_lds_v4i(void* p) {
  return (AS3 v4i*)(unsigned int)(unsigned long long)p;
}
#endif

// ---- WMMA fragment load from LDS ------------------------------------------
// 16-bit A/B 16x32 layout (ISA 7.12.2):
//   lane<16 : row=lane,    halves 0..7 -> K 0..7,  halves 8..15 -> K 16..23
//   lane>=16: row=lane-16, halves 0..7 -> K 8..15, halves 8..15 -> K 24..31
__device__ __forceinline__ v16h frag_load(const half_t* base, int row0, int lane) {
  const int r  = row0 + (lane & 15);
  const int kb = (lane >> 4) << 3;                   // 0 or 8
  const half_t* p = base + r * LDSS + kb;
  v8h lo = *(const v8h*)(p);
  v8h hi = *(const v8h*)(p + 16);
  return __builtin_shufflevector(lo, hi, 0,1,2,3,4,5,6,7,8,9,10,11,12,13,14,15);
}

// EP: 0 = f16 out, no bias, no act          (Ax)
//     1 = f16 out, bias + SiLU              (phi hidden h)
//     2 = f32 out, bias                     (phi, psi, G pre-LN)
template<int EP>
__device__ __forceinline__ void store_tile(void* outp, const float* __restrict__ bias,
                                           v8f c, int mrow0, int ncol0, int N, int lane) {
  const int col   = ncol0 + (lane & 15);
  const int rbase = mrow0 + ((lane >> 4) << 3);
  if (col >= N) return;                               // after all WMMAs; safe
  float bv = 0.f;
  if (EP != 0) bv = bias[col];
#pragma unroll
  for (int i = 0; i < 8; ++i) {
    float v = c[i] + bv;
    if (EP == 1) v = v / (1.f + __expf(-v));          // SiLU
    if (EP == 2) ((float*)outp)[(size_t)(rbase + i) * N + col] = v;
    else         ((half_t*)outp)[(size_t)(rbase + i) * N + col] = (half_t)v;
  }
}

// ---- stage one 256x32 A tile + 64x32 B tile into LDS ----------------------
// Async path: exactly 5 ASYNCcnt-tracked instructions per wave per tile
// (B row clamped in-bounds instead of predicated, so the count is constant).
__device__ __forceinline__ void stage_tile(const half_t* __restrict__ X,
                                           const half_t* __restrict__ Wt,
                                           half_t* sA, half_t* sB,
                                           int m0, int n0, int k0,
                                           int K, int N, int tid) {
#if USE_ASYNC
#pragma unroll
  for (int cc = 0; cc < 4; ++cc) {
    int ch = tid + cc * 256;          // 1024 16B chunks of A
    int r  = ch >> 2;
    int kk = (ch & 3) << 3;
    const half_t* gp = X + (size_t)(m0 + r) * K + (k0 + kk);
    __builtin_amdgcn_global_load_async_to_lds_b128(
        as_global_v4i(gp), as_lds_v4i(sA + r * LDSS + kk), 0, 0);
  }
  {
    int r  = tid >> 2;                // 256 16B chunks of B
    int kk = (tid & 3) << 3;
    int rowg = n0 + r; if (rowg > N - 1) rowg = N - 1;   // clamp: garbage cols unused
    const half_t* gp = Wt + (size_t)rowg * K + (k0 + kk);
    __builtin_amdgcn_global_load_async_to_lds_b128(
        as_global_v4i(gp), as_lds_v4i(sB + r * LDSS + kk), 0, 0);
  }
#else
#pragma unroll
  for (int cc = 0; cc < 4; ++cc) {
    int ch = tid + cc * 256;
    int r  = ch >> 2;
    int kk = (ch & 3) << 3;
    v8h v = *(const v8h*)(X + (size_t)(m0 + r) * K + (k0 + kk));
    *(v8h*)(sA + r * LDSS + kk) = v;
  }
  {
    int r  = tid >> 2;
    int kk = (tid & 3) << 3;
    int rowg = n0 + r; if (rowg > N - 1) rowg = N - 1;
    v8h v = *(const v8h*)(Wt + (size_t)rowg * K + (k0 + kk));
    *(v8h*)(sB + r * LDSS + kk) = v;
  }
#endif
}

// ---- 64x32 wave tile: 4 A frags x 2 B frags -> 8 WMMA ---------------------
__device__ __forceinline__ void compute_tile(const half_t* sA, const half_t* sB,
                                             int wm, int wn, int lane, v8f (&acc)[4][2]) {
  v16h a0 = frag_load(sA, wm +  0, lane);
  v16h a1 = frag_load(sA, wm + 16, lane);
  v16h a2 = frag_load(sA, wm + 32, lane);
  v16h a3 = frag_load(sA, wm + 48, lane);
  v16h b0 = frag_load(sB, wn +  0, lane);
  v16h b1 = frag_load(sB, wn + 16, lane);
  acc[0][0] = __builtin_amdgcn_wmma_f32_16x16x32_f16(false, a0, false, b0, (short)0, acc[0][0], false, false);
  acc[0][1] = __builtin_amdgcn_wmma_f32_16x16x32_f16(false, a0, false, b1, (short)0, acc[0][1], false, false);
  acc[1][0] = __builtin_amdgcn_wmma_f32_16x16x32_f16(false, a1, false, b0, (short)0, acc[1][0], false, false);
  acc[1][1] = __builtin_amdgcn_wmma_f32_16x16x32_f16(false, a1, false, b1, (short)0, acc[1][1], false, false);
  acc[2][0] = __builtin_amdgcn_wmma_f32_16x16x32_f16(false, a2, false, b0, (short)0, acc[2][0], false, false);
  acc[2][1] = __builtin_amdgcn_wmma_f32_16x16x32_f16(false, a2, false, b1, (short)0, acc[2][1], false, false);
  acc[3][0] = __builtin_amdgcn_wmma_f32_16x16x32_f16(false, a3, false, b0, (short)0, acc[3][0], false, false);
  acc[3][1] = __builtin_amdgcn_wmma_f32_16x16x32_f16(false, a3, false, b1, (short)0, acc[3][1], false, false);
}

// ---- Tiled WMMA GEMM: out = act(X[M,K] * Wt[N,K]^T + bias) ----------------
// X row-major f16, Wt = W^T row-major f16 ([N][K]). M%256==0, K%32==0.
template<int EP>
__global__ __launch_bounds__(256)
void gemm_wmma(const half_t* __restrict__ X, const half_t* __restrict__ Wt,
               const float* __restrict__ bias, void* __restrict__ outp,
               int M, int N, int K) {
  __shared__ half_t sA[2][BM * LDSS];
  __shared__ half_t sB[2][BN * LDSS];
  const int tid  = threadIdx.x;
  const int lane = tid & 31;
  const int wave = tid >> 5;
  const int m0 = blockIdx.y * BM;
  const int n0 = blockIdx.x * BN;
  const int wm = (wave >> 1) * 64;     // 4 waves along M (64 rows each)
  const int wn = (wave & 1) * 32;      // 2 waves along N (32 cols each)

  v8f acc[4][2] = {};
  const int nk = K / BK;

  stage_tile(X, Wt, sA[0], sB[0], m0, n0, 0, K, N, tid);
  for (int i = 0; i < nk; ++i) {
    const int cur = i & 1;
    if (i + 1 < nk) {
      stage_tile(X, Wt, sA[cur ^ 1], sB[cur ^ 1], m0, n0, (i + 1) * BK, K, N, tid);
#if USE_ASYNC
      __builtin_amdgcn_s_wait_asynccnt(5);   // keep next tile's 5 loads in flight
#endif
    } else {
#if USE_ASYNC
      __builtin_amdgcn_s_wait_asynccnt(0);
#endif
    }
    __syncthreads();
    compute_tile(sA[cur], sB[cur], wm, wn, lane, acc);
    __syncthreads();
  }
#pragma unroll
  for (int j = 0; j < 4; ++j) {
    store_tile<EP>(outp, bias, acc[j][0], m0 + wm + 16 * j, n0 + wn,      N, lane);
    store_tile<EP>(outp, bias, acc[j][1], m0 + wm + 16 * j, n0 + wn + 16, N, lane);
  }
}

// ---- W [K,N] f32 -> Wt [N,K] f16 (one t, pre-offset) ----------------------
__global__ __launch_bounds__(256)
void transpose_cast(const float* __restrict__ W, half_t* __restrict__ Wt, int K, int N) {
  __shared__ float tile[32][33];
  const int kb = blockIdx.x * 32, nb = blockIdx.y * 32;
  const int tx = threadIdx.x & 31, ty = threadIdx.x >> 5;   // 32x8
#pragma unroll
  for (int i = ty; i < 32; i += 8) {
    int k = kb + i, n = nb + tx;
    tile[i][tx] = (k < K && n < N) ? W[(size_t)k * N + n] : 0.f;
  }
  __syncthreads();
#pragma unroll
  for (int i = ty; i < 32; i += 8) {
    int n = nb + i, k = kb + tx;
    if (n < N && k < K) Wt[(size_t)n * K + k] = (half_t)tile[tx][i];
  }
}

// ---- Patch conv (patch P): y[b*np+p, c] f16 -------------------------------
template<int P>
__global__ __launch_bounds__(256)
void conv_patch(const float* __restrict__ x, const float* __restrict__ w,
                const float* __restrict__ bias, half_t* __restrict__ y, int nside) {
  int idx = blockIdx.x * blockDim.x + threadIdx.x;
  int total = 8 * nside * nside * 384;
  if (idx >= total) return;
  int c    = idx % 384;
  int rest = idx / 384;
  int p    = rest % (nside * nside);
  int b    = rest / (nside * nside);
  int ph = p / nside, pw = p % nside;
  float acc = bias[c];
#pragma unroll
  for (int cin = 0; cin < 3; ++cin)
#pragma unroll
    for (int kh = 0; kh < P; ++kh)
#pragma unroll
      for (int kw = 0; kw < P; ++kw)
        acc += x[((b * 3 + cin) * 64 + ph * P + kh) * 64 + pw * P + kw]
             * w[((c * 3 + cin) * P + kh) * P + kw];
  y[idx] = (half_t)acc;
}

// ---- LayerNorm + exact GELU, one wave32 per row ---------------------------
__global__ __launch_bounds__(256)
void ln_gelu(const float* __restrict__ G, const float* __restrict__ gamma,
             const float* __restrict__ beta, half_t* __restrict__ out,
             int Mrows, int D) {
  int wave = blockIdx.x * 8 + (threadIdx.x >> 5);
  int lane = threadIdx.x & 31;
  if (wave >= Mrows) return;
  const float* row = G + (size_t)wave * D;
  float s = 0.f, ss = 0.f;
  for (int j = lane; j < D; j += 32) { float v = row[j]; s += v; ss += v * v; }
#pragma unroll
  for (int off = 16; off; off >>= 1) {
    s  += __shfl_xor(s,  off, 32);
    ss += __shfl_xor(ss, off, 32);
  }
  float mean = s / D;
  float var  = ss / D - mean * mean;
  float rstd = rsqrtf(var + 1e-5f);
  half_t* orow = out + (size_t)wave * D;
  for (int j = lane; j < D; j += 32) {
    float v = (row[j] - mean) * rstd * gamma[j] + beta[j];
    orow[j] = (half_t)(v * 0.5f * (1.f + erff(v * 0.70710678118f)));
  }
}

// ---- per-t combine into concat buffer (branch) or d_out (fusion) ----------
__global__ __launch_bounds__(256)
void combine_branch(const float* __restrict__ phi, const float* __restrict__ psi,
                    const float* __restrict__ mix, const float* __restrict__ temp,
                    int t, half_t* __restrict__ zh, int np, int o, int colofs) {
  int idx = blockIdx.x * blockDim.x + threadIdx.x;
  int total = 8 * np * o;
  if (idx >= total) return;
  int j = idx % o, m = idx / o;
  int b = m / np, n = m % np;
  float v = phi[idx] * psi[idx] * temp[0] * mix[t];
  zh[((size_t)(b * 1024 + n)) * 768 + colofs + t * o + j] = (half_t)v;
}

__global__ __launch_bounds__(256)
void combine_fusion(const float* __restrict__ phi, const float* __restrict__ psi,
                    const float* __restrict__ mix, const float* __restrict__ temp,
                    int t, const float* __restrict__ pos, float* __restrict__ out, int o) {
  int idx = blockIdx.x * blockDim.x + threadIdx.x;
  int total = 8192 * o;
  if (idx >= total) return;
  int j = idx % o, m = idx / o;
  int n = m % 1024;
  int col = t * o + j;
  out[(size_t)m * 768 + col] = phi[idx] * psi[idx] * temp[0] * mix[t] + pos[n * 768 + col];
}

__global__ __launch_bounds__(256)
void zero_h8(half_t* __restrict__ p, int total8) {
  int idx = blockIdx.x * blockDim.x + threadIdx.x;
  if (idx < total8) { v8h z = {}; *(v8h*)(p + (size_t)idx * 8) = z; }
}

// ---------------------------------------------------------------------------
// Host-side orchestration
// ---------------------------------------------------------------------------
struct KaBufs {
  half_t *bufAx, *bufH, *bufG2;
  float  *bufG, *bufPhi, *bufPsi;
  half_t *wA, *wP1, *wP2, *wS1, *wS2;
};

static void run_ka(void* const* d_in, int base, const half_t* Xh,
                   int T, int din, int h, int o, int M,
                   bool fusion_mode, int np, int colofs,
                   half_t* zh, float* out, const float* pos,
                   const KaBufs& wsb, hipStream_t s) {
  const float* A_    = (const float*)d_in[base + 0];
  const float* pw1   = (const float*)d_in[base + 1];
  const float* pb1   = (const float*)d_in[base + 2];
  const float* pw2   = (const float*)d_in[base + 3];
  const float* pb2   = (const float*)d_in[base + 4];
  const float* sw1   = (const float*)d_in[base + 5];
  const float* sb1   = (const float*)d_in[base + 6];
  const float* lng   = (const float*)d_in[base + 7];
  const float* lnb   = (const float*)d_in[base + 8];
  const float* sw2   = (const float*)d_in[base + 9];
  const float* sb2   = (const float*)d_in[base + 10];
  const float* mix   = (const float*)d_in[base + 11];
  const float* temp  = (const float*)d_in[base + 12];

  const dim3 blk(256);
  for (int t = 0; t < T; ++t) {
    // Ax = X @ A[t]                                      -> f16 [M,din]
    transpose_cast<<<dim3(din / 32, din / 32), blk, 0, s>>>(A_ + (size_t)t * din * din, wsb.wA, din, din);
    gemm_wmma<0><<<dim3(din / 64, M / 256), blk, 0, s>>>(Xh, wsb.wA, nullptr, wsb.bufAx, M, din, din);
    // H = silu(X @ phi_w1[t] + phi_b1[t])                -> f16 [M,h]
    transpose_cast<<<dim3(din / 32, h / 32), blk, 0, s>>>(pw1 + (size_t)t * din * h, wsb.wP1, din, h);
    gemm_wmma<1><<<dim3(h / 64, M / 256), blk, 0, s>>>(Xh, wsb.wP1, pb1 + (size_t)t * h, wsb.bufH, M, h, din);
    // phi = H @ phi_w2[t] + phi_b2[t]                    -> f32 [M,o]
    transpose_cast<<<dim3(h / 32, o / 32), blk, 0, s>>>(pw2 + (size_t)t * h * o, wsb.wP2, h, o);
    gemm_wmma<2><<<dim3((o + 63) / 64, M / 256), blk, 0, s>>>(wsb.bufH, wsb.wP2, pb2 + (size_t)t * o, wsb.bufPhi, M, o, h);
    // G = Ax @ psi_w1[t] + psi_b1[t]                     -> f32 [M,din]
    transpose_cast<<<dim3(din / 32, din / 32), blk, 0, s>>>(sw1 + (size_t)t * din * din, wsb.wS1, din, din);
    gemm_wmma<2><<<dim3(din / 64, M / 256), blk, 0, s>>>(wsb.bufAx, wsb.wS1, sb1 + (size_t)t * din, wsb.bufG, M, din, din);
    // G2 = gelu(layernorm(G))                            -> f16 [M,din]
    ln_gelu<<<dim3((M + 7) / 8), blk, 0, s>>>(wsb.bufG, lng + (size_t)t * din, lnb + (size_t)t * din, wsb.bufG2, M, din);
    // psi = G2 @ psi_w2[t] + psi_b2[t]                   -> f32 [M,o]
    transpose_cast<<<dim3(din / 32, o / 32), blk, 0, s>>>(sw2 + (size_t)t * din * o, wsb.wS2, din, o);
    gemm_wmma<2><<<dim3((o + 63) / 64, M / 256), blk, 0, s>>>(wsb.bufG2, wsb.wS2, sb2 + (size_t)t * o, wsb.bufPsi, M, o, din);
    // combine
    if (!fusion_mode) {
      int total = 8 * np * o;
      combine_branch<<<dim3((total + 255) / 256), blk, 0, s>>>(wsb.bufPhi, wsb.bufPsi, mix, temp, t, zh, np, o, colofs);
    } else {
      int total = 8192 * o;
      combine_fusion<<<dim3((total + 255) / 256), blk, 0, s>>>(wsb.bufPhi, wsb.bufPsi, mix, temp, t, pos, out, o);
    }
  }
}

extern "C" void kernel_launch(void* const* d_in, const int* in_sizes, int n_in,
                              void* d_out, int out_size, void* d_ws, size_t ws_size,
                              hipStream_t stream) {
  (void)in_sizes; (void)n_in; (void)out_size;
  const float* x       = (const float*)d_in[0];
  const float* conv2_w = (const float*)d_in[1];
  const float* conv2_b = (const float*)d_in[2];
  const float* conv4_w = (const float*)d_in[3];
  const float* conv4_b = (const float*)d_in[4];
  const int KA2 = 5, KA4 = 18, FUS = 31;
  const float* pos     = (const float*)d_in[44];

  // workspace carve (all 256B aligned)
  char* wp = (char*)d_ws;
  auto carve = [&](size_t bytes) -> void* {
    void* p = (void*)wp; wp += (bytes + 255) & ~(size_t)255; return p;
  };
  half_t* y2h   = (half_t*)carve(8192ull * 384 * 2);
  half_t* y4h   = (half_t*)carve(2048ull * 384 * 2);
  half_t* zh    = (half_t*)carve(8192ull * 768 * 2);
  KaBufs wsb;
  wsb.bufAx  = (half_t*)carve(8192ull * 768 * 2);
  wsb.bufH   = (half_t*)carve(8192ull * 384 * 2);
  wsb.bufG2  = (half_t*)carve(8192ull * 768 * 2);
  wsb.bufG   = (float*) carve(8192ull * 768 * 4);
  wsb.bufPhi = (float*) carve(8192ull * 96 * 4);
  wsb.bufPsi = (float*) carve(8192ull * 96 * 4);
  wsb.wA     = (half_t*)carve(768ull * 768 * 2);
  wsb.wP1    = (half_t*)carve(384ull * 768 * 2);
  wsb.wP2    = (half_t*)carve(96ull * 384 * 2);
  wsb.wS1    = (half_t*)carve(768ull * 768 * 2);
  wsb.wS2    = (half_t*)carve(96ull * 768 * 2);
  if ((size_t)(wp - (char*)d_ws) > ws_size) return;   // insufficient scratch

  const dim3 blk(256);
  // 0) zero the padded concat buffer zh [8192,768] f16
  zero_h8<<<dim3(8192 * 768 / 8 / 256), blk, 0, stream>>>(zh, 8192 * 768 / 8);
  // 1) patch convs -> f16 row-major token matrices
  conv_patch<2><<<dim3(8 * 1024 * 384 / 256), blk, 0, stream>>>(x, conv2_w, conv2_b, y2h, 32);
  conv_patch<4><<<dim3(8 * 256 * 384 / 256), blk, 0, stream>>>(x, conv4_w, conv4_b, y4h, 16);
  // 2) branch KA layers -> zh columns [0,384) and [384,768)
  run_ka(d_in, KA2, y2h, 6, 384, 192, 64, 8192, false, 1024, 0,   zh, nullptr, nullptr, wsb, stream);
  run_ka(d_in, KA4, y4h, 6, 384, 192, 64, 2048, false, 256,  384, zh, nullptr, nullptr, wsb, stream);
  // 3) fusion KA layer -> d_out (f32) + pos_embed
  run_ka(d_in, FUS, zh, 8, 768, 384, 96, 8192, true, 1024, 0, nullptr, (float*)d_out, pos, wsb, stream);
}
